// GraphConvMapperBlock_34084860461562
// MI455X (gfx1250) — matrix operand order repaired and checked by reading, not compiled, not run on hardware
//
#include <hip/hip_runtime.h>
#include <hip/hip_bf16.h>

typedef _Float16 v16h __attribute__((ext_vector_type(16)));
typedef _Float16 v8h  __attribute__((ext_vector_type(8)));
typedef float    v8f  __attribute__((ext_vector_type(8)));

#define WMMA16(a, b, c) \
    __builtin_amdgcn_wmma_f32_16x16x32_f16(false, (a), false, (b), (short)0, (c), false, false)

__device__ __forceinline__ v8h cvt8(float4 f0, float4 f1) {
    v8h r;
    r[0] = (_Float16)f0.x; r[1] = (_Float16)f0.y; r[2] = (_Float16)f0.z; r[3] = (_Float16)f0.w;
    r[4] = (_Float16)f1.x; r[5] = (_Float16)f1.y; r[6] = (_Float16)f1.z; r[7] = (_Float16)f1.w;
    return r;
}

// Load a 16x32 f16 A-fragment for lane L from an LDS row-major tile.
// ISA 16-bit A layout: lane holds row m=L%16; K-half (L/16)*8 in dwords 0..3 (K=half*8..+7)
// and 16+half*8..+7 in dwords 4..7 -> two contiguous 16-byte LDS loads.
__device__ __forceinline__ v16h load_afrag(const _Float16* rowbase, int koff) {
    union { v16h v; v8h h[2]; } u;
    u.h[0] = *(const v8h*)(rowbase + koff);
    u.h[1] = *(const v8h*)(rowbase + koff + 16);
    return u.v;
}

// ---------------------------------------------------------------------------
// Swizzle fp32 weights [K,128] into f16 WMMA B-fragments.
// Fragment (kt,nt) stored as 32 lanes x 16 halves contiguous (1024B/frag).
// B layout: lane L -> n = nt*16 + L%16 ; k = kt*32 + (L/16)*8 + (h%8) + (h/8)*16.
// ---------------------------------------------------------------------------
__global__ void wswizzle_kernel(const float* __restrict__ w, _Float16* __restrict__ dst,
                                int ktiles) {
    int t = blockIdx.x * 256 + threadIdx.x;
    int total = ktiles * 8 * 512;
    if (t >= total) return;
    int fi = t >> 9, rem = t & 511;
    int L = rem >> 4, h = rem & 15;
    int v = h >> 1, lo = h & 1;
    int kt = fi >> 3, nt = fi & 7;
    int k = kt * 32 + (L >> 4) * 8 + (h & 7) + (h >> 3) * 16;
    int n = nt * 16 + (L & 15);
    dst[t] = (_Float16)w[(size_t)k * 128 + n];
}

// ---------------------------------------------------------------------------
// Edge MLP: edges_new = LN(silu(cat[x_dst[d],x_src[s],ea] @ W1 + b1) @ W2 + b2)
// plus segment-sum (atomic) into agg[d].
// One block = 16 edges; 8 waves, wave w computes output cols [16w,16w+16).
// ---------------------------------------------------------------------------
__global__ void __launch_bounds__(256) edge_mlp_kernel(
    const float* __restrict__ xsrc, const float* __restrict__ xdst,
    const float* __restrict__ eattr,
    const int* __restrict__ sidx, const int* __restrict__ didx, int E,
    const _Float16* __restrict__ w1f, const _Float16* __restrict__ w2f,
    const float* __restrict__ b1, const float* __restrict__ b2,
    const float* __restrict__ g, const float* __restrict__ bn,
    float* __restrict__ edges_out, float* __restrict__ agg) {
    __shared__ _Float16 A1[16 * 384];  // msg_in tile (f16)
    __shared__ _Float16 H[16 * 128];   // hidden (f16)
    __shared__ float    Y[16 * 128];   // pre-LN output
    __shared__ float    MU[16], RS[16];
    __shared__ int      DSTI[16];

    const int tid = threadIdx.x;
    const int e0 = blockIdx.x * 16;

    // ---- gather: 16 threads per row, 8 floats per segment each ----
    {
        const int row = tid >> 4, l16 = tid & 15;
        int e = e0 + row;
        int ec = (e < E) ? e : (E - 1);
        int s = sidx[ec], d = didx[ec];
        if (l16 == 0) DSTI[row] = d;
        const float4* pd = (const float4*)(xdst + (size_t)d * 128) + l16 * 2;
        const float4* ps = (const float4*)(xsrc + (size_t)s * 128) + l16 * 2;
        const float4* pe = (const float4*)(eattr + (size_t)ec * 128) + l16 * 2;
        _Float16* a = A1 + row * 384;
        *(v8h*)(a + 0 * 128 + l16 * 8) = cvt8(pd[0], pd[1]);
        *(v8h*)(a + 1 * 128 + l16 * 8) = cvt8(ps[0], ps[1]);
        *(v8h*)(a + 2 * 128 + l16 * 8) = cvt8(pe[0], pe[1]);
    }
    __syncthreads();

    const int w = tid >> 5;         // wave id == N-tile
    const int L = tid & 31;         // lane
    const int m = L & 15;           // A row held by this lane
    const int half = (L >> 4) * 8;  // K sub-offset
    const int n16 = L & 15;         // output column within tile

    // ---- GEMM1: [16x384] @ [384x128], K-tiles of 32 ----
    v8f c = {};
    const _Float16* arow = A1 + m * 384;
#pragma unroll
    for (int kt = 0; kt < 12; ++kt) {
        v16h a = load_afrag(arow, kt * 32 + half);
        v16h b = *(const v16h*)(w1f + ((size_t)(kt * 8 + w) * 32 + L) * 16);
        c = WMMA16(a, b, c);
    }
    // bias + SiLU, stage hidden as f16
    {
        float bias = b1[w * 16 + n16];
#pragma unroll
        for (int r = 0; r < 8; ++r) {
            float x = c[r] + bias;
            x = x / (1.0f + __expf(-x));
            int mm = (L < 16) ? r : (r + 8);
            H[mm * 128 + w * 16 + n16] = (_Float16)x;
        }
    }
    __syncthreads();

    // ---- GEMM2: [16x128] @ [128x128] ----
    v8f c2 = {};
    const _Float16* hrow = H + m * 128;
#pragma unroll
    for (int kt = 0; kt < 4; ++kt) {
        v16h a = load_afrag(hrow, kt * 32 + half);
        v16h b = *(const v16h*)(w2f + ((size_t)(kt * 8 + w) * 32 + L) * 16);
        c2 = WMMA16(a, b, c2);
    }
    {
        float bias = b2[w * 16 + n16];
#pragma unroll
        for (int r = 0; r < 8; ++r) {
            int mm = (L < 16) ? r : (r + 8);
            Y[mm * 128 + w * 16 + n16] = c2[r] + bias;
        }
    }
    __syncthreads();

    // ---- LayerNorm stats (one thread per row, staggered to dodge bank conflicts) ----
    if (tid < 16) {
        float s = 0.f, ss = 0.f;
        const float* yr = Y + tid * 128;
        for (int jj = 0; jj < 128; ++jj) {
            int i = (jj + tid * 8) & 127;
            float v = yr[i];
            s += v; ss += v * v;
        }
        float mu = s * (1.0f / 128.0f);
        float var = ss * (1.0f / 128.0f) - mu * mu;
        MU[tid] = mu;
        RS[tid] = rsqrtf(var + 1e-5f);
    }
    __syncthreads();

    // ---- apply LN, write edges_new, atomic segment-sum ----
    {
        const int row = tid >> 4, col0 = (tid & 15) * 8;
        int e = e0 + row;
        if (e < E) {
            float mu = MU[row], rs = RS[row];
            int d = DSTI[row];
            float* eo = edges_out + (size_t)e * 128 + col0;
            float* ag = agg + (size_t)d * 128 + col0;
#pragma unroll
            for (int j = 0; j < 8; ++j) {
                int col = col0 + j;
                float v = (Y[row * 128 + col] - mu) * rs * g[col] + bn[col];
                eo[j] = v;
                atomicAdd(&ag[j], v);
            }
        }
    }
}

// ---------------------------------------------------------------------------
// Node MLP: out = LN(silu(cat[x, second] @ W1 + b1) @ W2 + b2) + x
// second == agg (dst path) or x itself (src path).
// ---------------------------------------------------------------------------
__global__ void __launch_bounds__(256) node_mlp_kernel(
    const float* __restrict__ x, const float* __restrict__ second, int N,
    const _Float16* __restrict__ w1f, const _Float16* __restrict__ w2f,
    const float* __restrict__ b1, const float* __restrict__ b2,
    const float* __restrict__ g, const float* __restrict__ bn,
    float* __restrict__ out) {
    __shared__ _Float16 A1[16 * 256];
    __shared__ _Float16 H[16 * 128];
    __shared__ float    Y[16 * 128];
    __shared__ float    MU[16], RS[16];

    const int tid = threadIdx.x;
    const int n0 = blockIdx.x * 16;

    {
        const int row = tid >> 4, l16 = tid & 15;
        int node = n0 + row;
        int nc = (node < N) ? node : (N - 1);
        const float4* p0 = (const float4*)(x + (size_t)nc * 128) + l16 * 2;
        const float4* p1 = (const float4*)(second + (size_t)nc * 128) + l16 * 2;
        _Float16* a = A1 + row * 256;
        *(v8h*)(a + 0 * 128 + l16 * 8) = cvt8(p0[0], p0[1]);
        *(v8h*)(a + 1 * 128 + l16 * 8) = cvt8(p1[0], p1[1]);
    }
    __syncthreads();

    const int w = tid >> 5;
    const int L = tid & 31;
    const int m = L & 15;
    const int half = (L >> 4) * 8;
    const int n16 = L & 15;

    v8f c = {};
    const _Float16* arow = A1 + m * 256;
#pragma unroll
    for (int kt = 0; kt < 8; ++kt) {
        v16h a = load_afrag(arow, kt * 32 + half);
        v16h b = *(const v16h*)(w1f + ((size_t)(kt * 8 + w) * 32 + L) * 16);
        c = WMMA16(a, b, c);
    }
    {
        float bias = b1[w * 16 + n16];
#pragma unroll
        for (int r = 0; r < 8; ++r) {
            float xv = c[r] + bias;
            xv = xv / (1.0f + __expf(-xv));
            int mm = (L < 16) ? r : (r + 8);
            H[mm * 128 + w * 16 + n16] = (_Float16)xv;
        }
    }
    __syncthreads();

    v8f c2 = {};
    const _Float16* hrow = H + m * 128;
#pragma unroll
    for (int kt = 0; kt < 4; ++kt) {
        v16h a = load_afrag(hrow, kt * 32 + half);
        v16h b = *(const v16h*)(w2f + ((size_t)(kt * 8 + w) * 32 + L) * 16);
        c2 = WMMA16(a, b, c2);
    }
    {
        float bias = b2[w * 16 + n16];
#pragma unroll
        for (int r = 0; r < 8; ++r) {
            int mm = (L < 16) ? r : (r + 8);
            Y[mm * 128 + w * 16 + n16] = c2[r] + bias;
        }
    }
    __syncthreads();

    if (tid < 16) {
        float s = 0.f, ss = 0.f;
        const float* yr = Y + tid * 128;
        for (int jj = 0; jj < 128; ++jj) {
            int i = (jj + tid * 8) & 127;
            float v = yr[i];
            s += v; ss += v * v;
        }
        float mu = s * (1.0f / 128.0f);
        float var = ss * (1.0f / 128.0f) - mu * mu;
        MU[tid] = mu;
        RS[tid] = rsqrtf(var + 1e-5f);
    }
    __syncthreads();

    {
        const int row = tid >> 4, col0 = (tid & 15) * 8;
        int node = n0 + row;
        if (node < N) {
            float mu = MU[row], rs = RS[row];
            const float* xr = x + (size_t)node * 128 + col0;
            float* o = out + (size_t)node * 128 + col0;
#pragma unroll
            for (int j = 0; j < 8; ++j) {
                int col = col0 + j;
                float v = (Y[row * 128 + col] - mu) * rs * g[col] + bn[col];
                o[j] = v + xr[j];
            }
        }
    }
}

extern "C" void kernel_launch(void* const* d_in, const int* in_sizes, int n_in,
                              void* d_out, int out_size, void* d_ws, size_t ws_size,
                              hipStream_t stream) {
    const float* x_src  = (const float*)d_in[0];
    const float* x_dst  = (const float*)d_in[1];
    const float* eattr  = (const float*)d_in[2];
    const int*   eidx   = (const int*)d_in[3];
    const float* ew1 = (const float*)d_in[4];
    const float* eb1 = (const float*)d_in[5];
    const float* ew2 = (const float*)d_in[6];
    const float* eb2 = (const float*)d_in[7];
    const float* eg  = (const float*)d_in[8];
    const float* ebn = (const float*)d_in[9];
    const float* nw1 = (const float*)d_in[10];
    const float* nb1 = (const float*)d_in[11];
    const float* nw2 = (const float*)d_in[12];
    const float* nb2 = (const float*)d_in[13];
    const float* ng  = (const float*)d_in[14];
    const float* nbn = (const float*)d_in[15];

    const int C     = 128;
    const int NSRC  = in_sizes[0] / C;
    const int NDST  = in_sizes[1] / C;
    const int E     = in_sizes[3] / 2;
    const int* sidx = eidx;          // edge_index[0] = src
    const int* didx = eidx + E;      // edge_index[1] = dst

    // K-tile counts (K/32)
    const int KT_EW1 = (in_sizes[4] / C) / 32;   // 384/32 = 12
    const int KT_EW2 = (in_sizes[6] / C) / 32;   // 128/32 = 4
    const int KT_NW1 = (in_sizes[10] / C) / 32;  // 256/32 = 8
    const int KT_NW2 = (in_sizes[12] / C) / 32;  // 128/32 = 4

    // Workspace layout: [agg f32 NDST*128][ew1f][ew2f][nw1f][nw2f]
    float* agg = (float*)d_ws;
    size_t agg_bytes = (size_t)NDST * C * sizeof(float);
    _Float16* ew1f = (_Float16*)((char*)d_ws + agg_bytes);
    _Float16* ew2f = ew1f + (size_t)KT_EW1 * 8 * 512;
    _Float16* nw1f = ew2f + (size_t)KT_EW2 * 8 * 512;
    _Float16* nw2f = nw1f + (size_t)KT_NW1 * 8 * 512;

    // zero segment-sum accumulator every call (capture-safe memset node)
    hipMemsetAsync(agg, 0, agg_bytes, stream);

    // weight swizzle fp32 -> f16 B-fragments
    auto swz = [&](const float* src, _Float16* dst, int kt) {
        int total = kt * 8 * 512;
        wswizzle_kernel<<<(total + 255) / 256, 256, 0, stream>>>(src, dst, kt);
    };
    swz(ew1, ew1f, KT_EW1);
    swz(ew2, ew2f, KT_EW2);
    swz(nw1, nw1f, KT_NW1);
    swz(nw2, nw2f, KT_NW2);

    // outputs: [nodes_new_src | nodes_new_dst | edges_new]
    float* out_src   = (float*)d_out;
    float* out_dst   = out_src + (size_t)NSRC * C;
    float* edges_out = out_dst + (size_t)NDST * C;

    edge_mlp_kernel<<<(E + 15) / 16, 256, 0, stream>>>(
        x_src, x_dst, eattr, sidx, didx, E,
        ew1f, ew2f, eb1, eb2, eg, ebn, edges_out, agg);

    node_mlp_kernel<<<(NDST + 15) / 16, 256, 0, stream>>>(
        x_dst, agg, NDST, nw1f, nw2f, nb1, nb2, ng, nbn, out_dst);

    node_mlp_kernel<<<(NSRC + 15) / 16, 256, 0, stream>>>(
        x_src, x_src, NSRC, nw1f, nw2f, nb1, nb2, ng, nbn, out_src);
}
